// GNN_nl_68891275428276
// MI455X (gfx1250) — matrix operand order repaired and checked by reading, not compiled, not run on hardware
//
#include <hip/hip_runtime.h>
#include <cstdint>

typedef _Float16 h16;
typedef __attribute__((ext_vector_type(8)))  _Float16 v8h;
typedef __attribute__((ext_vector_type(16))) _Float16 v16h;
typedef __attribute__((ext_vector_type(8)))  short    v8s;
typedef __attribute__((ext_vector_type(8)))  float    v8f;

// ---------------------------------------------------------------------------
// Weight f32 [Nc][K] -> f16 padded [NcP][Kp] (zero pad), NcP mult 16, Kp mult 64
// ---------------------------------------------------------------------------
__global__ void convert_w_kernel(const float* __restrict__ w, h16* __restrict__ o,
                                 int Nc, int K, int Kp, int total) {
  int i = blockIdx.x * 256 + threadIdx.x;
  if (i >= total) return;
  int n = i / Kp, k = i % Kp;
  o[i] = (n < Nc && k < K) ? (h16)w[n * K + k] : (h16)0.f;
}

// x f32 [rows][K] -> f16 [rows][ldo] (pad columns already zeroed by memset)
__global__ void cvt_x_kernel(const float* __restrict__ x, h16* __restrict__ o,
                             int rows, int K, int ldo) {
  long i = (long)blockIdx.x * 256 + threadIdx.x;
  if (i >= (long)rows * K) return;
  int k = (int)(i % K);
  long r = i / K;
  o[r * ldo + k] = (h16)x[i];
}

__global__ void copy_cols_kernel(const h16* __restrict__ in, int ldin,
                                 h16* __restrict__ out, int ldout, int rows, int C) {
  long i = (long)blockIdx.x * 256 + threadIdx.x;
  if (i >= (long)rows * C) return;
  int c = (int)(i % C);
  long r = i / C;
  out[r * ldout + c] = in[r * ldin + c];
}

// xt[b][c][j] = x[b*128 + j][c]   (per-batch transpose, c < 320, j < 128)
__global__ void transpose_x_kernel(const h16* __restrict__ x, h16* __restrict__ xt,
                                   int ldx) {
  long i = (long)blockIdx.x * 256 + threadIdx.x;   // over 16*320*128
  if (i >= (long)16 * 320 * 128) return;
  int j = (int)(i & 127);
  int c = (int)((i >> 7) % 320);
  int b = (int)(i / (320 * 128));
  xt[i] = x[((size_t)b * 128 + j) * ldx + c];
}

__device__ __forceinline__ v16h make_afrag(const h16* ar, int hi) {
  v8h a0 = *(const v8h*)(ar + (hi << 3));
  v8h a1 = *(const v8h*)(ar + 16 + (hi << 3));
  return __builtin_shufflevector(a0, a1, 0, 1, 2, 3, 4, 5, 6, 7,
                                 8, 9, 10, 11, 12, 13, 14, 15);
}

// ---------------------------------------------------------------------------
// K-loop body, fully specialized on MODE (launch-uniform) and TWO (block-
// uniform, manually unswitched) so the hot loop carries no scalar branches.
// ---------------------------------------------------------------------------
template <int MODE, bool TWO>
__device__ __forceinline__ void gemm_kloop(
    const h16* __restrict__ xsrc, unsigned xbatch, unsigned xi0, int ldx,
    const h16* __restrict__ A, int lda,
    const h16* __restrict__ Wb, int ldw,
    h16* sA, int Kp, int m0, int nct0,
    int tid, int wave, int hi, int lo16, int kk8,
    v8f& acc00, v8f& acc01, v8f& acc10, v8f& acc11)
{
  for (int k0 = 0; k0 < Kp; k0 += 64) {
    __syncthreads();
    if (MODE == 0) {
      v8h xiA = *(const v8h*)(xsrc + (xi0 + (unsigned)(k0 + kk8)));
      v8h xiB = *(const v8h*)(xsrc + (xi0 + (unsigned)ldx + (unsigned)(k0 + kk8)));
#pragma unroll
      for (int t = 0; t < 8; ++t) {
        int r = t * 32 + (tid >> 3);              // 0..255
        int j = r & 127;
        v8h xi = (t < 4) ? xiA : xiB;             // compile-time per t
        v8h xj = *(const v8h*)(xsrc + (xbatch + (unsigned)j * (unsigned)ldx +
                                       (unsigned)(k0 + kk8)));
        v8h dsub = xi - xj;                       // v_pk_sub_f16
        v8s su = (v8s)dsub;
        su &= (short)0x7fff;                      // abs via sign mask
        *(v8h*)(sA + r * 64 + kk8) = (v8h)su;
      }
    } else {
#pragma unroll
      for (int t = 0; t < 8; ++t) {
        int r = t * 32 + (tid >> 3);
        *(v8h*)(sA + r * 64 + kk8) =
            *(const v8h*)(A + ((unsigned)(m0 + r) * (unsigned)lda +
                               (unsigned)(k0 + kk8)));
      }
      if (k0 + 64 < Kp)                           // warm next panel
        __builtin_prefetch(A + ((unsigned)(m0 + tid) * (unsigned)lda +
                                (unsigned)(k0 + 64)), 0, 1);
    }
    __syncthreads();

#pragma unroll
    for (int kk = 0; kk < 64; kk += 32) {
      // A fragments for two M-tiles (ISA A 16x32 striping = 2x contiguous 8-half)
      const h16* ar0 = sA + ((wave << 5) + lo16) * 64 + kk;
      v16h af0 = make_afrag(ar0, hi);
      v16h af1 = make_afrag(ar0 + 16 * 64, hi);

      unsigned wb0 = (unsigned)((nct0 << 4) + lo16) * (unsigned)ldw +
                     (unsigned)(k0 + kk);
      v16h bf0 = make_afrag(Wb + wb0, hi);
      acc00 = __builtin_amdgcn_wmma_f32_16x16x32_f16(false, af0, false, bf0,
                                                     (short)0, acc00, false, false);
      acc10 = __builtin_amdgcn_wmma_f32_16x16x32_f16(false, af1, false, bf0,
                                                     (short)0, acc10, false, false);
      if (TWO) {
        v16h bf1 = make_afrag(Wb + wb0 + 16u * (unsigned)ldw, hi);
        acc01 = __builtin_amdgcn_wmma_f32_16x16x32_f16(false, af0, false, bf1,
                                                       (short)0, acc01, false, false);
        acc11 = __builtin_amdgcn_wmma_f32_16x16x32_f16(false, af1, false, bf1,
                                                       (short)0, acc11, false, false);
      }
    }
  }
}

// ---------------------------------------------------------------------------
// Unified WMMA GEMM:  C[M x Nc] = A[M x K] * W^T + bias   (W stored [NcP][Kp])
//   MODE 0: A row (b,i,j) = |x[b,i,:] - x[b,j,:]| built on the fly (f16)
//   MODE 1: A read from global f16 [M][lda]   (lda >= roundup64(K), pad zeroed)
//   wbstride!=0: W batched per 128-row group (waves 0-3 batch0, 4-7 batch1)
// grid = (M/256, ceil(NcP/32)), block = 256 (8 waves); each wave computes
// TWO 16-row M-tiles x TWO 16-col N-tiles -> 8 v_wmma per 256x64 LDS panel.
// ---------------------------------------------------------------------------
template <int MODE>
__global__ __launch_bounds__(256) void gemm_wmma_kernel(
    const h16* __restrict__ xsrc, int ldx,        // MODE 0 source
    const h16* __restrict__ A,  int lda,          // MODE 1 source
    const h16* __restrict__ W,  int ldw, long wbstride, int rpb,
    const float* __restrict__ bias,
    h16* __restrict__ outh, float* __restrict__ outf, int ldc, int act,
    int M, int Nc, int K)
{
  __shared__ h16 sA[256 * 64];
  const int tid  = threadIdx.x;
  const int wave = tid >> 5;
  const int lane = tid & 31;
  const int hi   = lane >> 4;
  const int lo16 = lane & 15;
  const int m0   = blockIdx.x * 256;
  const int nct0 = blockIdx.y * 2;
  const int NcP  = (Nc + 15) & ~15;
  const bool two = (nct0 + 1) * 16 < NcP;         // uniform across block
  const int Kp   = (K + 63) & ~63;

  const h16* Wb = W;
  if (wbstride) Wb = W + (size_t)(m0 / rpb) * wbstride
                       + (size_t)(wave >> 2) * wbstride;  // 2nd 128-row group

  // MODE 0 decode: m = ((b*128 + i)*128 + j); 256 rows = (i0, i0+1) x j(0..127)
  unsigned xbatch = 0, xi0 = 0;
  if (MODE == 0) {
    unsigned b  = (unsigned)(m0 >> 14);
    unsigned i0 = (unsigned)((m0 >> 7) & 127);
    xbatch = b * 128u * (unsigned)ldx;            // batch row base
    xi0    = xbatch + i0 * (unsigned)ldx;         // row i0 base
  }

  const int kk8 = (tid & 7) * 8;                  // 16B column slot in panel
  v8f acc00 = {}, acc01 = {}, acc10 = {}, acc11 = {};

  if (two)
    gemm_kloop<MODE, true >(xsrc, xbatch, xi0, ldx, A, lda, Wb, ldw, sA, Kp,
                            m0, nct0, tid, wave, hi, lo16, kk8,
                            acc00, acc01, acc10, acc11);
  else
    gemm_kloop<MODE, false>(xsrc, xbatch, xi0, ldx, A, lda, Wb, ldw, sA, Kp,
                            m0, nct0, tid, wave, hi, lo16, kk8,
                            acc00, acc01, acc10, acc11);

  // ---- store (f32 C/D layout: VGPR r -> row r + 8*hi, col = lane&15) ----
  const int rbase = m0 + (wave << 5);
  {
    int n = (nct0 << 4) + lo16;
    if (n < Nc) {
      float bv = bias ? bias[n] : 0.f;
#pragma unroll
      for (int r = 0; r < 8; ++r) {
        size_t row0 = (size_t)(rbase + r + (hi << 3));
        float v0 = acc00[r] + bv, v1 = acc10[r] + bv;
        if (act == 1) { v0 = v0 > 0.f ? v0 : expm1f(v0); v1 = v1 > 0.f ? v1 : expm1f(v1); }
        if (outh) { outh[row0 * ldc + n] = (h16)v0; outh[(row0 + 16) * ldc + n] = (h16)v1; }
        else      { outf[row0 * ldc + n] = v0;      outf[(row0 + 16) * ldc + n] = v1; }
      }
    }
  }
  if (two) {
    int n = ((nct0 + 1) << 4) + lo16;
    if (n < Nc) {
      float bv = bias ? bias[n] : 0.f;
#pragma unroll
      for (int r = 0; r < 8; ++r) {
        size_t row0 = (size_t)(rbase + r + (hi << 3));
        float v0 = acc01[r] + bv, v1 = acc11[r] + bv;
        if (act == 1) { v0 = v0 > 0.f ? v0 : expm1f(v0); v1 = v1 > 0.f ? v1 : expm1f(v1); }
        if (outh) { outh[row0 * ldc + n] = (h16)v0; outh[(row0 + 16) * ldc + n] = (h16)v1; }
        else      { outf[row0 * ldc + n] = v0;      outf[(row0 + 16) * ldc + n] = v1; }
      }
    }
  }
}

// ---------------------------------------------------------------------------
// Per-channel sum / sumsq over M rows of f16 [M][ld]; one block per channel.
// ---------------------------------------------------------------------------
__global__ __launch_bounds__(256) void stats_kernel(const h16* __restrict__ h, int ld,
                                                    int M, float* __restrict__ stats, int C) {
  int c = blockIdx.x;
  float s = 0.f, s2 = 0.f;
  for (int r = threadIdx.x; r < M; r += 256) {
    float v = (float)h[(size_t)r * ld + c];
    s += v; s2 += v * v;
  }
  __shared__ float rs[256], rq[256];
  rs[threadIdx.x] = s; rq[threadIdx.x] = s2;
  __syncthreads();
  for (int o = 128; o > 0; o >>= 1) {
    if (threadIdx.x < o) { rs[threadIdx.x] += rs[threadIdx.x + o]; rq[threadIdx.x] += rq[threadIdx.x + o]; }
    __syncthreads();
  }
  if (threadIdx.x == 0) { stats[c] = rs[0]; stats[C + c] = rq[0]; }
}

// BN (biased var) + activation, f16 in -> f16 out (act: 0 none, 1 elu, 2 leaky 0.01)
__global__ __launch_bounds__(256) void bn_act_kernel(const h16* __restrict__ in, int ldin,
                                                     h16* __restrict__ out, int ldout,
                                                     const float* __restrict__ stats,
                                                     const float* __restrict__ g,
                                                     const float* __restrict__ be,
                                                     int M, int C, int act) {
  long i = (long)blockIdx.x * 256 + threadIdx.x;
  if (i >= (long)M * C) return;
  int c = (int)(i % C);
  long r = i / C;
  float mean = stats[c] / (float)M;
  float var  = stats[C + c] / (float)M - mean * mean;
  float v = ((float)in[r * ldin + c] - mean) * rsqrtf(var + 1e-5f) * g[c] + be[c];
  if (act == 1)      v = v > 0.f ? v : expm1f(v);
  else if (act == 2) v = v > 0.f ? v : 0.01f * v;
  out[r * ldout + c] = (h16)v;
}

// logits[m] = blast + sum_k (h[m][k] + h04[m][k]) * wlast[k]   (96 -> 1 conv + skip)
__global__ __launch_bounds__(256) void wlast_dot_kernel(const h16* __restrict__ h,
                                                        const h16* __restrict__ h04,
                                                        const float* __restrict__ w,
                                                        const float* __restrict__ b,
                                                        float* __restrict__ logits, long M) {
  long m = (long)blockIdx.x * 256 + threadIdx.x;
  if (m >= M) return;
  const v8h* r1 = (const v8h*)(h + m * 96);
  const v8h* r2 = (const v8h*)(h04 + m * 96);
  float s = b[0];
#pragma unroll
  for (int v = 0; v < 12; ++v) {
    v8h a = r1[v], c = r2[v];
#pragma unroll
    for (int e = 0; e < 8; ++e) s += ((float)a[e] + (float)c[e]) * w[v * 8 + e];
  }
  logits[m] = s;
}

// masked softmax over j (128) per (b,i) row; one wave per row; writes f16 probs
__global__ __launch_bounds__(256) void softmax_kernel(const float* __restrict__ logits,
                                                      h16* __restrict__ s16, int BN) {
  int row  = blockIdx.x * 8 + (threadIdx.x >> 5);
  int lane = threadIdx.x & 31;
  if (row >= BN) return;
  int i = row & 127;
  const float* p = logits + (size_t)row * 128;
  float v[4], mx = -1e30f;
#pragma unroll
  for (int t = 0; t < 4; ++t) {
    int j = lane + 32 * t;
    float x = p[j];
    if (j == i) x -= 1e8f;
    v[t] = x;
    mx = fmaxf(mx, x);
  }
  for (int o = 16; o; o >>= 1) mx = fmaxf(mx, __shfl_xor(mx, o));
  float sum = 0.f;
#pragma unroll
  for (int t = 0; t < 4; ++t) { v[t] = __expf(v[t] - mx); sum += v[t]; }
  for (int o = 16; o; o >>= 1) sum += __shfl_xor(sum, o);
  float inv = 1.f / sum;
#pragma unroll
  for (int t = 0; t < 4; ++t) s16[(size_t)row * 128 + lane + 32 * t] = (h16)(v[t] * inv);
}

// symmetric distill-KL over N axis per (b,c); T=4 -> T*T/B == 1
__global__ __launch_bounds__(256) void kl_kernel(const h16* __restrict__ xs,
                                                 const h16* __restrict__ xt,
                                                 int ld, int C, float* __restrict__ mloss) {
  int w = blockIdx.x * 8 + (threadIdx.x >> 5);
  int lane = threadIdx.x & 31;
  if (w >= 16 * C) return;
  int b = w / C, c = w % C;
  const h16* ps = xs + (size_t)b * 128 * ld + c;
  const h16* pt = xt + (size_t)b * 128 * ld + c;
  float vs[4], vt[4], ms = -1e30f, mt = -1e30f;
#pragma unroll
  for (int t = 0; t < 4; ++t) {
    int n = lane + 32 * t;
    vs[t] = (float)ps[(size_t)n * ld] * 0.25f;
    vt[t] = (float)pt[(size_t)n * ld] * 0.25f;
    ms = fmaxf(ms, vs[t]); mt = fmaxf(mt, vt[t]);
  }
  for (int o = 16; o; o >>= 1) { ms = fmaxf(ms, __shfl_xor(ms, o)); mt = fmaxf(mt, __shfl_xor(mt, o)); }
  float ss = 0.f, st = 0.f;
#pragma unroll
  for (int t = 0; t < 4; ++t) { ss += __expf(vs[t] - ms); st += __expf(vt[t] - mt); }
  for (int o = 16; o; o >>= 1) { ss += __shfl_xor(ss, o); st += __shfl_xor(st, o); }
  float lzs = __logf(ss), lzt = __logf(st);
  float acc = 0.f;
#pragma unroll
  for (int t = 0; t < 4; ++t) {
    float lps = vs[t] - ms - lzs, lpt = vt[t] - mt - lzt;
    acc += __expf(lpt) * (lpt - lps) + __expf(lps) * (lps - lpt);
  }
  for (int o = 16; o; o >>= 1) acc += __shfl_xor(acc, o);
  if (lane == 0) atomicAdd(mloss, acc);
}

__global__ void add_out_kernel(const float* __restrict__ a, const float* __restrict__ b,
                               float* __restrict__ o, int n) {
  int i = blockIdx.x * 256 + threadIdx.x;
  if (i < n) o[i] = a[i] + b[i];
}

// ---------------------------------------------------------------------------
// Host orchestration
// ---------------------------------------------------------------------------
extern "C" void kernel_launch(void* const* d_in, const int* in_sizes, int n_in,
                              void* d_out, int out_size, void* d_ws, size_t ws_size,
                              hipStream_t stream)
{
  (void)in_sizes; (void)n_in; (void)out_size; (void)ws_size;
  const int N = 128, DIN = 128, NF = 96, HALF = 48, NWAY = 5;
  const long Mp = 262144;   // B*N*N
  const int  BN = 2048;     // B*N
  const int  LDX = 320;     // padded feature stride (>= roundup64(272))

  auto F = [&](int i) { return (const float*)d_in[i]; };

  size_t off = 0;
  auto alloc = [&](size_t bytes) -> void* {
    void* p = (char*)d_ws + off;
    off += (bytes + 255) & ~(size_t)255;
    return p;
  };

  h16*  xf    = (h16*) alloc((size_t)BN * LDX * 2);
  h16*  xof   = (h16*) alloc((size_t)BN * LDX * 2);
  h16*  xfT   = (h16*) alloc((size_t)16 * LDX * 128 * 2);   // [b][c][j]
  h16*  xofT  = (h16*) alloc((size_t)16 * LDX * 128 * 2);
  h16*  hA    = (h16*) alloc((size_t)Mp * 192 * 2);
  h16*  hB    = (h16*) alloc((size_t)Mp * 192 * 2);
  h16*  hC    = (h16*) alloc((size_t)Mp * 128 * 2);         // ld 128, pad zeroed
  h16*  h04   = (h16*) alloc((size_t)Mp * 96 * 2);
  float* logits = (float*)alloc((size_t)Mp * 4);
  h16*  sX    = (h16*) alloc((size_t)Mp * 2);
  h16*  sO    = (h16*) alloc((size_t)Mp * 2);
  h16*  gx    = (h16*) alloc((size_t)BN * 576 * 2);
  h16*  gxo   = (h16*) alloc((size_t)BN * 576 * 2);
  h16*  tmp48 = (h16*) alloc((size_t)BN * 48 * 2);
  float* outA = (float*)alloc((size_t)BN * NWAY * 4);
  float* outB = (float*)alloc((size_t)BN * NWAY * 4);
  float* stats= (float*)alloc(2 * 256 * 4);

  float* co    = (float*)d_out;
  float* mloss = co + (size_t)BN * NWAY;
  hipMemsetAsync(mloss, 0, sizeof(float), stream);

  // zero padded buffers so all GEMM reads past K hit zeros (branchless staging)
  hipMemsetAsync(xf,  0, (size_t)BN * LDX * 2, stream);
  hipMemsetAsync(xof, 0, (size_t)BN * LDX * 2, stream);
  hipMemsetAsync(hC,  0, (size_t)Mp * 128 * 2, stream);
  hipMemsetAsync(gx,  0, (size_t)BN * 576 * 2, stream);
  hipMemsetAsync(gxo, 0, (size_t)BN * 576 * 2, stream);

  auto cw = [&](const float* w, int Nc, int K) -> h16* {
    int NcP = (Nc + 15) & ~15, Kp = (K + 63) & ~63;
    int tot = NcP * Kp;
    h16* o = (h16*)alloc((size_t)tot * 2);
    convert_w_kernel<<<(tot + 255) / 256, 256, 0, stream>>>(w, o, Nc, K, Kp, tot);
    return o;
  };

  auto gemm = [&](int mode, const h16* xsrc, int ldx, const h16* A, int lda,
                  const h16* W, int ldw, long wbs, int rpb,
                  const float* bias, h16* outh, float* outf, int ldc,
                  int act, long M, int Nc, int K) {
    int NcP = (Nc + 15) & ~15;
    dim3 g((unsigned)(M / 256), (unsigned)((NcP + 31) / 32));
    if (mode == 0)
      gemm_wmma_kernel<0><<<g, 256, 0, stream>>>(xsrc, ldx, A, lda, W, ldw,
                                                 wbs, rpb, bias, outh, outf, ldc,
                                                 act, (int)M, Nc, K);
    else
      gemm_wmma_kernel<1><<<g, 256, 0, stream>>>(xsrc, ldx, A, lda, W, ldw,
                                                 wbs, rpb, bias, outh, outf, ldc,
                                                 act, (int)M, Nc, K);
  };

  auto bn = [&](const h16* in, int ldin, h16* out, int ldout,
                const float* g_, const float* be, long M, int C, int act) {
    stats_kernel<<<C, 256, 0, stream>>>(in, ldin, (int)M, stats, C);
    long tot = M * C;
    bn_act_kernel<<<(unsigned)((tot + 255) / 256), 256, 0, stream>>>(
        in, ldin, out, ldout, stats, g_, be, (int)M, C, act);
  };

  auto transpose = [&](const h16* x, h16* xt) {
    long tot = (long)16 * LDX * 128;
    transpose_x_kernel<<<(unsigned)((tot + 255) / 256), 256, 0, stream>>>(x, xt, LDX);
  };

  // one full Wcompute: x(f16,[B,N,d]) -> softmaxed adjacency sOut (f16,[B,N,N])
  auto wcompute = [&](const h16* xsrc, int d, int pb, h16* sOut) {
    const float *w04 = F(pb+0),  *b04 = F(pb+1),  *g04 = F(pb+2),  *be04 = F(pb+3);
    const float *w1  = F(pb+4),  *b1  = F(pb+5),  *g1  = F(pb+6),  *be1  = F(pb+7);
    const float *w2  = F(pb+8),  *b2  = F(pb+9),  *g2  = F(pb+10), *be2  = F(pb+11);
    const float *w3  = F(pb+12), *b3  = F(pb+13), *g3  = F(pb+14), *be3  = F(pb+15);
    const float *w4  = F(pb+16), *b4  = F(pb+17), *g4  = F(pb+18), *be4  = F(pb+19);
    const float *wl  = F(pb+20), *bl  = F(pb+21);
    int Kp = (d + 63) & ~63;
    h16 *w1h  = cw(w1, 2*NF, d), *w04h = cw(w04, NF, d),
        *w2h  = cw(w2, 2*NF, 2*NF), *w3h = cw(w3, NF, 2*NF), *w4h = cw(w4, NF, NF);

    gemm(0, xsrc, LDX, nullptr, 0, w1h, Kp, 0, 0, b1, hA, nullptr, 192, 0, Mp, 192, d);
    bn(hA, 192, hA, 192, g1, be1, Mp, 192, 1);                      // elu
    gemm(0, xsrc, LDX, nullptr, 0, w04h, Kp, 0, 0, b04, h04, nullptr, 96, 0, Mp, 96, d);
    bn(h04, 96, h04, 96, g04, be04, Mp, 96, 2);                     // leaky_relu
    gemm(1, nullptr, 0, hA, 192, w2h, 192, 0, 0, b2, hB, nullptr, 192, 0, Mp, 192, 192);
    bn(hB, 192, hB, 192, g2, be2, Mp, 192, 1);
    gemm(1, nullptr, 0, hB, 192, w3h, 192, 0, 0, b3, hC, nullptr, 128, 0, Mp, 96, 192);
    bn(hC, 128, hC, 128, g3, be3, Mp, 96, 1);
    gemm(1, nullptr, 0, hC, 128, w4h, 128, 0, 0, b4, hA, nullptr, 96, 0, Mp, 96, 96);
    bn(hA, 96, hA, 96, g4, be4, Mp, 96, 1);
    wlast_dot_kernel<<<(unsigned)((Mp + 255) / 256), 256, 0, stream>>>(hA, h04, wl, bl, logits, Mp);
    softmax_kernel<<<BN / 8, 256, 0, stream>>>(logits, sOut, BN);
  };

  // convert x (and x_ori = x) to f16 with padded stride
  {
    unsigned g = (unsigned)(((long)BN * DIN + 255) / 256);
    cvt_x_kernel<<<g, 256, 0, stream>>>((const float*)d_in[0], xf,  BN, DIN, LDX);
    cvt_x_kernel<<<g, 256, 0, stream>>>((const float*)d_in[0], xof, BN, DIN, LDX);
  }

  const long WBS = (long)LDX * 128;   // per-batch stride of transposed x
  int d = DIN;
  for (int i = 0; i < 3; ++i) {
    int pb = 1 + i * 28;
    wcompute(xf, d, pb, sX);
    const float *lw = F(pb+22), *lb = F(pb+23);
    const float *ow = F(pb+24), *ob = F(pb+25), *og = F(pb+26), *obe = F(pb+27);

    transpose(xf, xfT);
    transpose(xof, xofT);

    // x path: gx = [x , s@x]; xn = elu(gx @ lw^T + lb) -> concat into xf
    gemm(1, nullptr, 0, sX, 128, xfT, 128, WBS, 128,
         nullptr, gx + d, nullptr, 576, 0, BN, d, 128);
    copy_cols_kernel<<<(unsigned)(((long)BN * d + 255) / 256), 256, 0, stream>>>(xf, LDX, gx, 576, BN, d);
    h16* lwh = cw(lw, HALF, 2 * d);
    gemm(1, nullptr, 0, gx, 576, lwh, (2*d + 63) & ~63, 0, 0,
         lb, xf + d, nullptr, LDX, 1, BN, HALF, 2 * d);

    // ori path: gxo = [x_ori , s@x_ori]; BN1d + elu -> concat into xof
    gemm(1, nullptr, 0, sX, 128, xofT, 128, WBS, 128,
         nullptr, gxo + d, nullptr, 576, 0, BN, d, 128);
    copy_cols_kernel<<<(unsigned)(((long)BN * d + 255) / 256), 256, 0, stream>>>(xof, LDX, gxo, 576, BN, d);
    h16* owh = cw(ow, HALF, 2 * d);
    gemm(1, nullptr, 0, gxo, 576, owh, (2*d + 63) & ~63, 0, 0,
         ob, tmp48, nullptr, 48, 0, BN, HALF, 2 * d);
    bn(tmp48, 48, xof + d, LDX, og, obe, BN, 48, 1);

    d += HALF;
    kl_kernel<<<(unsigned)((16 * d + 7) / 8), 256, 0, stream>>>(xf, xof, LDX, d, mloss);
  }

  // final heads
  wcompute(xf,  d, 85, sX);
  wcompute(xof, d, 85, sO);
  const float *Lw = F(107), *Lb = F(108), *Ow = F(109), *Ob = F(110);

  transpose(xf, xfT);
  transpose(xof, xofT);

  gemm(1, nullptr, 0, sX, 128, xfT, 128, WBS, 128,
       nullptr, gx + d, nullptr, 576, 0, BN, d, 128);
  copy_cols_kernel<<<(unsigned)(((long)BN * d + 255) / 256), 256, 0, stream>>>(xf, LDX, gx, 576, BN, d);
  h16* Lwh = cw(Lw, NWAY, 2 * d);
  gemm(1, nullptr, 0, gx, 576, Lwh, (2*d + 63) & ~63, 0, 0,
       Lb, nullptr, outA, NWAY, 2, BN, NWAY, 2 * d);

  gemm(1, nullptr, 0, sO, 128, xofT, 128, WBS, 128,
       nullptr, gxo + d, nullptr, 576, 0, BN, d, 128);
  copy_cols_kernel<<<(unsigned)(((long)BN * d + 255) / 256), 256, 0, stream>>>(xof, LDX, gxo, 576, BN, d);
  h16* Owh = cw(Ow, NWAY, 2 * d);
  gemm(1, nullptr, 0, gxo, 576, Owh, (2*d + 63) & ~63, 0, 0,
       Ob, nullptr, outB, NWAY, 2, BN, NWAY, 2 * d);

  add_out_kernel<<<(BN * NWAY + 255) / 256, 256, 0, stream>>>(outA, outB, co, BN * NWAY);
}
